// LSTM_Seq2Dis_50156628083180
// MI455X (gfx1250) — compile-verified, matrix operand order checked by reading, not run on hardware
//
#include <hip/hip_runtime.h>
#include <cstddef>

// ---------------------------------------------------------------------------
// Problem constants (from reference): B=64, T=128, H=50, V=8192
// ---------------------------------------------------------------------------
#define BB   64
#define TT   128
#define HH   50
#define VV   8192
#define MM   (BB * (TT / 2))   // 4096
#define G4   (4 * HH)          // 200

typedef __attribute__((ext_vector_type(16))) _Float16 v16h;
typedef __attribute__((ext_vector_type(8)))  _Float16 v8h;
typedef __attribute__((ext_vector_type(8)))  float    v8f;
typedef int v4i __attribute__((vector_size(16)));   // matches builtin prototype

union hv16 { v16h v; v8h h8[2]; };

// ---------------------------------------------------------------------------
// CDNA5 async global->LDS copy (16B per call) + asynccnt wait.
// Builtin signature (from hipcc diagnostic): (AS1 v4i*, AS3 v4i*, imm, imm).
// AS3 pointer value = low 32 bits of the generic LDS address (wave-relative).
// ---------------------------------------------------------------------------
__device__ __forceinline__ void async_cp16(const void* g, void* l) {
    unsigned int loff = (unsigned int)(unsigned long long)l;
#if __has_builtin(__builtin_amdgcn_global_load_async_to_lds_b128)
    __builtin_amdgcn_global_load_async_to_lds_b128(
        (__attribute__((address_space(1))) v4i*)(unsigned long long)g,
        (__attribute__((address_space(3))) v4i*)loff, 0, 0);
#else
    asm volatile("global_load_async_to_lds_b128 %0, %1, off"
                 :: "v"(loff), "v"(g) : "memory");
#endif
}

__device__ __forceinline__ void wait_async0() {
#if __has_builtin(__builtin_amdgcn_s_wait_asynccnt)
    __builtin_amdgcn_s_wait_asynccnt(0);
#else
    asm volatile("s_wait_asynccnt 0" ::: "memory");
#endif
}

// ---------------------------------------------------------------------------
// Kernel 1: embedding gather + input-projection GEMM (tiny: K=50)
//   zx[t][b][g*50+k] = b4[g*50+k] + sum_h E[idx[b,t],h] * Wx[h, g*50+k]
// ---------------------------------------------------------------------------
__global__ __launch_bounds__(256)
void embed_zx_kernel(const int* __restrict__ idx, const float* __restrict__ E,
                     const float* __restrict__ Wx, const float* __restrict__ b4,
                     float* __restrict__ zx) {
    int o = blockIdx.x * 256 + threadIdx.x;
    if (o >= TT * BB * G4) return;
    int n = o % G4;
    int b = (o / G4) % BB;
    int t = o / (G4 * BB);
    int tok = idx[b * TT + t];
    const float* e = &E[(size_t)tok * HH];
    float acc = b4[n];
#pragma unroll 10
    for (int h = 0; h < HH; ++h) acc += e[h] * Wx[(size_t)h * G4 + n];
    zx[o] = acc;
}

// ---------------------------------------------------------------------------
// Kernel 2: LSTM recurrence + fused MaxPool1D(2).  One workgroup per batch
// element; U (40KB), masks, h/c state all in LDS.
// ---------------------------------------------------------------------------
__global__ __launch_bounds__(256)
void lstm_pool_kernel(const float* __restrict__ zx, const float* __restrict__ U,
                      const float* __restrict__ rmask, float* __restrict__ pooled) {
    const int b   = blockIdx.x;
    const int tid = threadIdx.x;

    __shared__ float Us[4 * HH * HH];
    __shared__ float mk[G4];
    __shared__ float hs[HH], cs[HH], ph[HH];
    __shared__ float hm[G4];
    __shared__ float zb[G4];

    for (int i = tid; i < 4 * HH * HH; i += 256) Us[i] = U[i];
    for (int i = tid; i < G4; i += 256) {
        int g = i / HH, h = i % HH;
        mk[i] = rmask[((size_t)g * BB + b) * HH + h];
    }
    if (tid < HH) { hs[tid] = 0.f; cs[tid] = 0.f; ph[tid] = 0.f; }
    __syncthreads();

    for (int t = 0; t < TT; ++t) {
        if (tid < G4) hm[tid] = hs[tid % HH] * mk[tid];
        __syncthreads();
        if (tid < G4) {
            int g = tid / HH, k = tid % HH;
            float acc = zx[((size_t)t * BB + b) * G4 + tid];
            const float* ug  = &Us[g * HH * HH + k];
            const float* hmg = &hm[g * HH];
#pragma unroll 10
            for (int h = 0; h < HH; ++h) acc += hmg[h] * ug[h * HH];
            zb[tid] = acc;
        }
        __syncthreads();
        if (tid < HH) {
            float zi = zb[tid], zf = zb[HH + tid], zg = zb[2 * HH + tid], zo = zb[3 * HH + tid];
            float ig = 1.f / (1.f + __expf(-zi));
            float fg = 1.f / (1.f + __expf(-zf));
            float gg = tanhf(zg);
            float og = 1.f / (1.f + __expf(-zo));
            float cn = fg * cs[tid] + ig * gg;
            float hn = og * tanhf(cn);
            cs[tid] = cn; hs[tid] = hn;
            if (t & 1) pooled[((size_t)b * (TT / 2) + (t >> 1)) * HH + tid] = fmaxf(ph[tid], hn);
            else       ph[tid] = hn;
        }
        __syncthreads();
    }
}

// ---------------------------------------------------------------------------
// Kernel 3: dense1 (K=50): x1 = f16(relu(pooled@W1+b1)*dmask1); W1 L2-resident.
// ---------------------------------------------------------------------------
__global__ __launch_bounds__(256)
void dense1_kernel(const float* __restrict__ pooled, const float* __restrict__ W1,
                   const float* __restrict__ b1, const float* __restrict__ dmask1,
                   _Float16* __restrict__ x1) {
    const int m = blockIdx.y;
    const int n = blockIdx.x * 256 + threadIdx.x;
    __shared__ float p[HH];
    if (threadIdx.x < HH) p[threadIdx.x] = pooled[(size_t)m * HH + threadIdx.x];
    __syncthreads();
    float acc = b1[n];
#pragma unroll 10
    for (int k = 0; k < HH; ++k) acc += p[k] * W1[(size_t)k * VV + n];
    size_t o = (size_t)m * VV + n;
    acc = fmaxf(acc, 0.f) * dmask1[o];
    x1[o] = (_Float16)acc;
}

// ---------------------------------------------------------------------------
// Kernels 4/5: 4096x8192x8192 GEMMs via v_wmma_f32_16x16x32_f16.
//   A (f16 activations): async global->LDS DMA (ASYNCcnt), no register trip.
//   B (fp32 weights):    register double-buffered stream, f32->f16 convert,
//                        transposed (n-major) LDS store.
//   LDS double buffered; one barrier per K-tile; prologue primes tile 0.
//   MODE 0: out = f16(relu(acc+bias) * mask)   MODE 1: out = f32(sigmoid(acc+bias))
// ---------------------------------------------------------------------------
template <int MODE>
__global__ __launch_bounds__(256)
void gemm_big_kernel(const _Float16* __restrict__ A, const float* __restrict__ Bw,
                     const float* __restrict__ bias, const float* __restrict__ mask,
                     _Float16* __restrict__ outH, float* __restrict__ outF) {
    constexpr int K = VV, N = VV, NK = VV / 32;
    const int tid  = threadIdx.x;
    const int lane = tid & 31;
    const int wave = tid >> 5;
    const int wm   = wave & 3;
    const int wn   = wave >> 2;
    const int n0   = blockIdx.x * 128;
    const int m0   = blockIdx.y * 128;

    __shared__ __align__(16) _Float16 As[2][128][32];   // row-major  [m][k]
    __shared__ __align__(16) _Float16 Bt[2][128][32];   // transposed [n][k]

    v8f acc[2][4];
#pragma unroll
    for (int i = 0; i < 2; ++i)
#pragma unroll
        for (int j = 0; j < 4; ++j)
#pragma unroll
            for (int e = 0; e < 8; ++e) acc[i][j][e] = 0.f;

    const int klo  = (lane >> 4) * 8;   // per-lane K-half (ISA frag layout)
    const int rsel = lane & 15;
    const int moff = (lane >> 4) * 8;   // C/D: lanes 16-31 hold M = v+8

    float4 breg[4];                      // in-flight B tile (fp32)

    // ---- prologue: prime tile 0 ----
#pragma unroll
    for (int j = 0; j < 2; ++j) {        // A: 512 x 16B async chunks
        int q = tid + j * 256;
        int r = q >> 2, c = (q & 3) * 8;
        async_cp16(&A[(size_t)(m0 + r) * K + c], &As[0][r][c]);
    }
#pragma unroll
    for (int j = 0; j < 4; ++j) {        // B: 4 float4 loads per thread
        int q = tid + j * 256;
        int kr = q >> 5, cg = (q & 31) * 4;
        breg[j] = *(const float4*)&Bw[(size_t)kr * N + n0 + cg];
    }

    for (int kt = 0; kt < NK; ++kt) {
        const int sb = kt & 1;

        // ---- complete staging of tile kt ----
#pragma unroll
        for (int j = 0; j < 4; ++j) {    // convert + transposed store of B
            int q = tid + j * 256;
            int kr = q >> 5, cg = (q & 31) * 4;
            Bt[sb][cg + 0][kr] = (_Float16)breg[j].x;
            Bt[sb][cg + 1][kr] = (_Float16)breg[j].y;
            Bt[sb][cg + 2][kr] = (_Float16)breg[j].z;
            Bt[sb][cg + 3][kr] = (_Float16)breg[j].w;
        }
        wait_async0();                   // A tile landed in LDS
        __syncthreads();

        // ---- issue staging for tile kt+1 (overlaps compute below) ----
        if (kt + 1 < NK) {
            const int kk = (kt + 1) * 32;
#pragma unroll
            for (int j = 0; j < 2; ++j) {
                int q = tid + j * 256;
                int r = q >> 2, c = (q & 3) * 8;
                async_cp16(&A[(size_t)(m0 + r) * K + kk + c], &As[sb ^ 1][r][c]);
            }
#pragma unroll
            for (int j = 0; j < 4; ++j) {
                int q = tid + j * 256;
                int kr = q >> 5, cg = (q & 31) * 4;
                breg[j] = *(const float4*)&Bw[(size_t)(kk + kr) * N + n0 + cg];
            }
        }

        // ---- compute on buffer sb: 8 WMMAs ----
        v16h af[2];
#pragma unroll
        for (int fm = 0; fm < 2; ++fm) {
            int r = wm * 32 + fm * 16 + rsel;
            hv16 u;
            u.h8[0] = *(const v8h*)&As[sb][r][klo];
            u.h8[1] = *(const v8h*)&As[sb][r][klo + 16];
            af[fm] = u.v;
        }
#pragma unroll
        for (int fn = 0; fn < 4; ++fn) {
            int cN = wn * 64 + fn * 16 + rsel;
            hv16 u;
            u.h8[0] = *(const v8h*)&Bt[sb][cN][klo];
            u.h8[1] = *(const v8h*)&Bt[sb][cN][klo + 16];
            v16h bf = u.v;
#pragma unroll
            for (int fm = 0; fm < 2; ++fm)
                acc[fm][fn] = __builtin_amdgcn_wmma_f32_16x16x32_f16(
                    false, af[fm], false, bf, (short)0, acc[fm][fn], false, false);
        }
    }

    // ---- fused epilogue ----
#pragma unroll
    for (int fm = 0; fm < 2; ++fm)
#pragma unroll
        for (int fn = 0; fn < 4; ++fn) {
            int nn = n0 + wn * 64 + fn * 16 + rsel;
            float bn = bias[nn];
#pragma unroll
            for (int v = 0; v < 8; ++v) {
                int mm = m0 + wm * 32 + fm * 16 + moff + v;
                size_t o = (size_t)mm * N + nn;
                float val = acc[fm][fn][v] + bn;
                if (MODE == 0) {
                    val = fmaxf(val, 0.f) * mask[o];
                    outH[o] = (_Float16)val;
                } else {
                    outF[o] = 1.f / (1.f + __expf(-val));
                }
            }
        }
}

// ---------------------------------------------------------------------------
// Launcher
// ---------------------------------------------------------------------------
extern "C" void kernel_launch(void* const* d_in, const int* in_sizes, int n_in,
                              void* d_out, int out_size, void* d_ws, size_t ws_size,
                              hipStream_t stream) {
    (void)in_sizes; (void)n_in; (void)out_size; (void)ws_size;

    const int*   bi    = (const int*)d_in[0];    // [B,T]
    const float* E     = (const float*)d_in[1];  // [V,H]
    const float* Wx    = (const float*)d_in[2];  // [H,4H]
    const float* U     = (const float*)d_in[3];  // [4,H,H]
    const float* b4    = (const float*)d_in[4];  // [4H]
    const float* W1    = (const float*)d_in[5];  // [H,V]
    const float* b1    = (const float*)d_in[6];  // [V]
    const float* W1b   = (const float*)d_in[7];  // [V,V]
    const float* W2    = (const float*)d_in[8];  // [V,V]
    const float* b2    = (const float*)d_in[9];  // [V]
    const float* rmask = (const float*)d_in[10]; // [4,B,H]
    const float* dm1   = (const float*)d_in[11]; // [B,T/2,V]
    const float* dm2   = (const float*)d_in[12]; // [B,T/2,V]
    float* out = (float*)d_out;                  // [B,T/2,V]

    const size_t ZX_ELEMS   = (size_t)TT * BB * G4;
    const size_t POOL_ELEMS = (size_t)MM * HH;
    const size_t X_ELEMS    = (size_t)MM * VV;
    float*     zx     = (float*)d_ws;
    float*     pooled = zx + ZX_ELEMS;
    _Float16*  x1     = (_Float16*)(pooled + POOL_ELEMS);
    _Float16*  x2     = x1 + X_ELEMS;

    embed_zx_kernel<<<(TT * BB * G4 + 255) / 256, 256, 0, stream>>>(bi, E, Wx, b4, zx);
    lstm_pool_kernel<<<BB, 256, 0, stream>>>(zx, U, rmask, pooled);
    dense1_kernel<<<dim3(VV / 256, MM), 256, 0, stream>>>(pooled, W1, b1, dm1, x1);
    gemm_big_kernel<0><<<dim3(VV / 128, MM / 128), 256, 0, stream>>>(
        x1, W1b, b1, dm2, x2, nullptr);
    gemm_big_kernel<1><<<dim3(VV / 128, MM / 128), 256, 0, stream>>>(
        x2, W2, b2, nullptr, nullptr, out);
}
